// EnhancedResGCN_14499809591446
// MI455X (gfx1250) — matrix operand order repaired and checked by reading, not compiled, not run on hardware
//
#include <hip/hip_runtime.h>
#include <hip/hip_bf16.h>

typedef __attribute__((ext_vector_type(16))) _Float16 v16h;
typedef __attribute__((ext_vector_type(8)))  float    v8f;

// ---------------------------------------------------------------------------
// Elementwise helpers
// ---------------------------------------------------------------------------
__global__ void fill_kernel(float* __restrict__ p, float v, long long n) {
  long long i = (long long)blockIdx.x * blockDim.x + threadIdx.x;
  if (i < n) p[i] = v;
}

__global__ void degree_kernel(const int* __restrict__ src, const int* __restrict__ dst,
                              float* __restrict__ od, float* __restrict__ idg, int E) {
  int e = blockIdx.x * blockDim.x + threadIdx.x;
  if (e >= E) return;
  atomicAdd(od + src[e], 1.0f);
  atomicAdd(idg + dst[e], 1.0f);
}

// in-place: od -> out_deg^-0.5, idg -> in_deg^-0.5, invid -> 1/in_deg (clamped)
__global__ void norm_kernel(float* od, float* idg, float* invid, int n) {
  int i = blockIdx.x * blockDim.x + threadIdx.x;
  if (i >= n) return;
  float o = fmaxf(od[i], 1.0f);
  float d = fmaxf(idg[i], 1.0f);
  od[i]    = rsqrtf(o);
  idg[i]   = rsqrtf(d);
  invid[i] = 1.0f / d;
}

// agg[dst] += h[src] * sscale[src] * dscale[dst]   (float4 chunks, scalar f32 atomics)
__global__ void scatter_add_kernel(const float* __restrict__ h,
                                   const float* __restrict__ sscale,
                                   const float* __restrict__ dscale,
                                   const int* __restrict__ src, const int* __restrict__ dst,
                                   float* __restrict__ agg, long long total, int Hc) {
  long long idx = (long long)blockIdx.x * blockDim.x + threadIdx.x;
  if (idx >= total) return;
  int chunks = Hc >> 2;
  int e = (int)(idx / chunks);
  int c = (int)(idx % chunks);
  int s = src[e], d = dst[e];
  const float4 v = *(const float4*)(h + (size_t)s * Hc + (size_t)c * 4);
  float sc = 1.0f;
  if (sscale) sc *= sscale[s];
  if (dscale) sc *= dscale[d];
  float* out = agg + (size_t)d * Hc + (size_t)c * 4;
  atomicAdd(out + 0, v.x * sc);
  atomicAdd(out + 1, v.y * sc);
  atomicAdd(out + 2, v.z * sc);
  atomicAdd(out + 3, v.w * sc);
}

// ---------------------------------------------------------------------------
// Weight pre-pack: f32 row-major W[K x NO] -> f16 B-fragment tiles.
// Tile t = tk * (NO/16) + tn holds 32 lanes x 16 halves; lane's half j is
// W[tk*32 + kbB + j][tn*16 + (lane&15)], kbB = (lane&16) ? 16 : 0.
// ---------------------------------------------------------------------------
__global__ void pack_w_kernel(const float* __restrict__ W, _Float16* __restrict__ Wp,
                              int K, int NO) {
  int idx = blockIdx.x * blockDim.x + threadIdx.x;
  int tilesN = NO >> 4;
  int total = (K >> 5) * tilesN * 32;
  if (idx >= total) return;
  int lane = idx & 31;
  int t    = idx >> 5;
  int tk   = t / tilesN;
  int tn   = t - tk * tilesN;
  int ncol = (tn << 4) + (lane & 15);
  int kbB  = (lane & 16) ? 16 : 0;
  _Float16* out = Wp + (size_t)t * 512 + (size_t)lane * 16;
#pragma unroll
  for (int j = 0; j < 16; ++j) {
    out[j] = (_Float16)W[(size_t)((tk << 5) + kbB + j) * NO + ncol];
  }
}

// ---------------------------------------------------------------------------
// WMMA GEMM:  C[M x NO] = act( rowscale(m) * A[M x K] @ W[K x NO] + bias )
// A split at compile-time KSPLIT: cols [0,KSPLIT) from A0, rest from A1.
// Wp is the f16 pre-packed B (pack_w_kernel layout). One wave = one 16x16
// tile; K-loop fully unrolled -> K/32 v_wmma_f32_16x16x32_f16 per tile.
// Requires: M%16==0, NO%16==0, K%32==0, KSPLIT%32==0.
// ---------------------------------------------------------------------------
template <int K, int NO, int KSPLIT>
__global__ void wmma_gemm_kernel(const float* __restrict__ A0, int lda0,
                                 const float* __restrict__ A1, int lda1,
                                 const float* __restrict__ rowscale,
                                 const _Float16* __restrict__ Wp,
                                 const float* __restrict__ bias,
                                 float* __restrict__ C, int ldc,
                                 int Mrows, int relu) {
  constexpr int tilesN = NO >> 4;
  const int lane = threadIdx.x & 31;
  const int wave = threadIdx.x >> 5;
  const int totalTiles = (Mrows >> 4) * tilesN;
  int t = blockIdx.x * (blockDim.x >> 5) + wave;
  if (t >= totalTiles) return;
  const int tm = t / tilesN;
  const int tn = t - tm * tilesN;

  const int mrow = (tm << 4) + (lane & 15);   // A row for this lane
  const int ncol = (tn << 4) + (lane & 15);   // C column for this lane
  const int kbA  = (lane & 16) ? 16 : 0;      // A K-base (high lanes +16)
  const float rs = rowscale ? rowscale[mrow] : 1.0f;

  const float* rowA0 = A0 + (size_t)mrow * lda0;
  const float* rowA1 = A1 + (size_t)mrow * lda1;

  v8f acc = {};
#pragma unroll
  for (int kt = 0; kt < K / 32; ++kt) {
    // ---- A fragment: contiguous k = 16h + s (pairs of even k) -------------
    v16h a;
#pragma unroll
    for (int v = 0; v < 8; ++v) {
      int k = (kt << 5) + kbA + (v << 1);
      float2 x;
      if (kt * 32 < KSPLIT) {                 // compile-time per unrolled kt
        x = *(const float2*)(rowA0 + k);
      } else {
        x = *(const float2*)(rowA1 + (k - KSPLIT));
      }
      a[2 * v]     = (_Float16)(x.x * rs);
      a[2 * v + 1] = (_Float16)(x.y * rs);
    }
    // ---- B fragment: one aligned 32B load from the packed tile -----------
    v16h b = *(const v16h*)(Wp + ((size_t)(kt * tilesN + tn) * 512) +
                            (size_t)lane * 16);
    acc = __builtin_amdgcn_wmma_f32_16x16x32_f16(false, a, false, b,
                                                 (short)0, acc, false, false);
  }

  const float bb = bias ? bias[ncol] : 0.0f;
  const int mbase = (tm << 4) + ((lane & 16) ? 8 : 0);  // C: VGPR r -> row mbase+r
#pragma unroll
  for (int r = 0; r < 8; ++r) {
    float v = acc[r] + bb;
    if (relu) v = fmaxf(v, 0.0f);
    C[(size_t)(mbase + r) * ldc + ncol] = v;
  }
}

template <int K, int NO, int KSPLIT>
static void launch_gemm(const float* A0, int lda0, const float* A1, int lda1,
                        const float* rowscale, const _Float16* Wp, const float* bias,
                        float* C, int ldc, int Mrows, int relu, hipStream_t stream) {
  int tiles  = (Mrows / 16) * (NO / 16);
  int blocks = (tiles + 3) / 4;  // 4 waves (128 threads) per block
  wmma_gemm_kernel<K, NO, KSPLIT><<<blocks, 128, 0, stream>>>(
      A0, lda0, A1, lda1, rowscale, Wp, bias, C, ldc, Mrows, relu);
}

// ---------------------------------------------------------------------------
// BatchNorm (H=64 features): stats, finalize, apply(+residual,+relu)
// ---------------------------------------------------------------------------
__global__ void bn_stats_kernel(const float* __restrict__ z, float* __restrict__ sums,
                                long long total) {
  // blockDim=256; grid-stride is a multiple of 64 -> each thread sticks to one feature
  float s = 0.0f, ss = 0.0f;
  long long stride = (long long)gridDim.x * blockDim.x;
  for (long long idx = (long long)blockIdx.x * blockDim.x + threadIdx.x; idx < total;
       idx += stride) {
    float v = z[idx];
    s += v; ss += v * v;
  }
  __shared__ float sh[512];
  sh[threadIdx.x]       = s;
  sh[256 + threadIdx.x] = ss;
  __syncthreads();
  if (threadIdx.x < 64) {
    float rs  = sh[threadIdx.x] + sh[threadIdx.x + 64] + sh[threadIdx.x + 128] + sh[threadIdx.x + 192];
    float rss = sh[256 + threadIdx.x] + sh[256 + threadIdx.x + 64] +
                sh[256 + threadIdx.x + 128] + sh[256 + threadIdx.x + 192];
    atomicAdd(sums + threadIdx.x, rs);
    atomicAdd(sums + 64 + threadIdx.x, rss);
  }
}

__global__ void bn_finalize_kernel(const float* __restrict__ sums, float* __restrict__ mv,
                                   float invN) {
  int f = threadIdx.x;  // 64 threads
  float mu  = sums[f] * invN;
  float var = sums[64 + f] * invN - mu * mu;
  mv[f]      = mu;
  mv[64 + f] = rsqrtf(var + 1e-5f);
}

__global__ void bn_apply_kernel(const float* __restrict__ z, const float* __restrict__ mv,
                                const float* __restrict__ gamma, const float* __restrict__ beta,
                                float* hout, const float* hres, long long total) {
  long long idx = (long long)blockIdx.x * blockDim.x + threadIdx.x;
  if (idx >= total) return;
  int f = (int)(idx & 63);
  float v = (z[idx] - mv[f]) * mv[64 + f] * gamma[f] + beta[f];
  if (hres) v += hres[idx];
  hout[idx] = fmaxf(v, 0.0f);
}

// ---------------------------------------------------------------------------
// Attention tail: a = sigmoid(T1 @ w2 + b2); then h += a * nb
// ---------------------------------------------------------------------------
__global__ void att_score_kernel(const float* __restrict__ t1, const float* __restrict__ w2,
                                 const float* __restrict__ b2, float* __restrict__ av, int n) {
  int i = blockIdx.x * blockDim.x + threadIdx.x;
  if (i >= n) return;
  float acc = 0.0f;
  const float* row = t1 + (size_t)i * 64;
#pragma unroll
  for (int f = 0; f < 64; ++f) acc += row[f] * w2[f];
  acc += b2[0];
  av[i] = 1.0f / (1.0f + __expf(-acc));
}

__global__ void h_update_kernel(float* __restrict__ h, const float* __restrict__ nb,
                                const float* __restrict__ av, long long total) {
  long long idx = (long long)blockIdx.x * blockDim.x + threadIdx.x;
  if (idx >= total) return;
  h[idx] += av[idx >> 6] * nb[idx];
}

// ---------------------------------------------------------------------------
// Host-side orchestration
// ---------------------------------------------------------------------------
static inline int cdiv(long long a, int b) { return (int)((a + b - 1) / b); }

extern "C" void kernel_launch(void* const* d_in, const int* in_sizes, int n_in,
                              void* d_out, int out_size, void* d_ws, size_t ws_size,
                              hipStream_t stream) {
  const float* features = (const float*)d_in[0];
  const int*   src      = (const int*)d_in[1];
  const int*   dst      = (const int*)d_in[2];
  const float* enc_W1   = (const float*)d_in[3];
  const float* enc_b1   = (const float*)d_in[4];
  const float* enc_W2   = (const float*)d_in[5];
  const float* enc_b2   = (const float*)d_in[6];
  const float* enc_W3   = (const float*)d_in[7];
  const float* enc_b3   = (const float*)d_in[8];
  const float* att_W1   = (const float*)d_in[9];
  const float* att_b1   = (const float*)d_in[10];
  const float* att_W2   = (const float*)d_in[11];
  const float* att_b2   = (const float*)d_in[12];
  const float* W0       = (const float*)d_in[13];
  const float* b0       = (const float*)d_in[14];
  const float* W_rest   = (const float*)d_in[15];
  const float* b_rest   = (const float*)d_in[16];
  const float* gamma    = (const float*)d_in[17];
  const float* beta     = (const float*)d_in[18];
  const float* fc_W     = (const float*)d_in[19];
  const float* fc_b     = (const float*)d_in[20];

  const int N = in_sizes[0] / 128;  // 100000
  const int E = in_sizes[1];        // 1600000
  const int L = 4;

  // ---- workspace carve-up (floats) ----
  float* ws = (float*)d_ws;
  size_t off = 0;
  auto alloc = [&](size_t nf) { float* p = ws + off; off += nf; return p; };
  float* outnorm  = alloc(N);
  float* innorm   = alloc(N);
  float* invindeg = alloc(N);
  float* H128     = alloc((size_t)N * 128);  // encoder output (layer-0 input)
  float* HCUR     = alloc((size_t)N * 64);   // current hidden h
  float* NB       = alloc((size_t)N * 64);   // neighbor mean
  float* AGG      = alloc((size_t)N * 128);  // SpMM accumulator
  float* T1       = alloc((size_t)N * 64);   // scratch (enc hidden 1 / att hidden)
  float* T2       = alloc((size_t)N * 32);   // scratch (enc hidden 2)
  float* AV       = alloc(N);                // attention scalar
  float* Z        = alloc((size_t)N * 64);   // conv output pre-BN
  float* STATS    = alloc(128);              // sum[64] + sumsq[64]
  float* MV       = alloc(128);              // mean[64] + invstd[64]

  // packed f16 weights (32B aligned: round up to 8-float boundary)
  off = (off + 7) & ~(size_t)7;
  auto allocH = [&](size_t halves) {        // halves is multiple of 512
    _Float16* p = (_Float16*)(ws + off);
    off += halves / 2;
    return p;
  };
  _Float16* P_enc1 = allocH(4 * 4 * 512);   // 128x64
  _Float16* P_enc2 = allocH(2 * 2 * 512);   // 64x32
  _Float16* P_enc3 = allocH(1 * 8 * 512);   // 32x128
  _Float16* P_att1 = allocH(4 * 4 * 512);   // 128x64
  _Float16* P_W0   = allocH(4 * 4 * 512);   // 128x64
  _Float16* P_Wr0  = allocH(2 * 4 * 512);   // 64x64
  _Float16* P_Wr1  = allocH(2 * 4 * 512);
  _Float16* P_Wr2  = allocH(2 * 4 * 512);
  _Float16* P_fc   = allocH(2 * 1 * 512);   // 64x16
  (void)ws_size; (void)n_in; (void)out_size;

  auto fill = [&](float* p, long long n) {
    fill_kernel<<<cdiv(n, 256), 256, 0, stream>>>(p, 0.0f, n);
  };
  auto pack = [&](const float* W, _Float16* Wp, int K, int NO) {
    int total = (K / 32) * (NO / 16) * 32;
    pack_w_kernel<<<cdiv(total, 256), 256, 0, stream>>>(W, Wp, K, NO);
  };
  auto scatter = [&](const float* h, const float* ss, const float* ds, float* agg, int Hc) {
    long long total = (long long)E * (Hc >> 2);
    scatter_add_kernel<<<cdiv(total, 256), 256, 0, stream>>>(h, ss, ds, src, dst, agg,
                                                             total, Hc);
  };

  // ---- pack all weights to f16 fragment layout ----
  pack(enc_W1, P_enc1, 128, 64);
  pack(enc_W2, P_enc2, 64, 32);
  pack(enc_W3, P_enc3, 32, 128);
  pack(att_W1, P_att1, 128, 64);
  pack(W0, P_W0, 128, 64);
  pack(W_rest + 0 * 64 * 64, P_Wr0, 64, 64);
  pack(W_rest + 1 * 64 * 64, P_Wr1, 64, 64);
  pack(W_rest + 2 * 64 * 64, P_Wr2, 64, 64);
  pack(fc_W, P_fc, 64, 16);
  _Float16* P_Wr[3] = {P_Wr0, P_Wr1, P_Wr2};

  // ---- degrees & norms ----
  fill(outnorm, N); fill(innorm, N);
  degree_kernel<<<cdiv(E, 256), 256, 0, stream>>>(src, dst, outnorm, innorm, E);
  norm_kernel<<<cdiv(N, 256), 256, 0, stream>>>(outnorm, innorm, invindeg, N);

  // ---- relation pattern encoder ----
  launch_gemm<128, 64, 128>(features, 128, features, 128, nullptr, P_enc1, enc_b1,
                            T1, 64, N, 1, stream);
  launch_gemm<64, 32, 64>(T1, 64, T1, 64, nullptr, P_enc2, enc_b2, T2, 32, N, 1, stream);
  launch_gemm<32, 128, 32>(T2, 32, T2, 32, nullptr, P_enc3, enc_b3, H128, 128, N, 0, stream);

  // ---- GCN layers ----
  for (int i = 0; i < L; ++i) {
    if (i >= 1) {
      // neighbor mean: NB[d] = (1/in_deg[d]) * sum_{e: dst=d} HCUR[src[e]]
      fill(NB, (long long)N * 64);
      scatter(HCUR, nullptr, invindeg, NB, 64);
      // attention gate on concat([h, nb]) (split-A GEMM, no materialized concat)
      launch_gemm<128, 64, 64>(HCUR, 64, NB, 64, nullptr, P_att1, att_b1, T1, 64, N, 1,
                               stream);
      att_score_kernel<<<cdiv(N, 256), 256, 0, stream>>>(T1, att_W2, att_b2, AV, N);
      h_update_kernel<<<cdiv((long long)N * 64, 256), 256, 0, stream>>>(
          HCUR, NB, AV, (long long)N * 64);
    }

    // graph_conv: AGG[d] = sum_e out_norm[s]*h[s]; Z = (in_norm * AGG) @ W + b
    if (i == 0) {
      fill(AGG, (long long)N * 128);
      scatter(H128, outnorm, nullptr, AGG, 128);
      launch_gemm<128, 64, 128>(AGG, 128, AGG, 128, innorm, P_W0, b0, Z, 64, N, 0,
                                stream);
    } else {
      fill(AGG, (long long)N * 64);
      scatter(HCUR, outnorm, nullptr, AGG, 64);
      launch_gemm<64, 64, 64>(AGG, 64, AGG, 64, innorm, P_Wr[i - 1],
                              b_rest + (size_t)(i - 1) * 64, Z, 64, N, 0, stream);
    }

    // batchnorm + residual + relu -> HCUR
    fill(STATS, 128);
    bn_stats_kernel<<<512, 256, 0, stream>>>(Z, STATS, (long long)N * 64);
    bn_finalize_kernel<<<1, 64, 0, stream>>>(STATS, MV, 1.0f / (float)N);
    bn_apply_kernel<<<cdiv((long long)N * 64, 256), 256, 0, stream>>>(
        Z, MV, gamma + (size_t)i * 64, beta + (size_t)i * 64, HCUR,
        (i > 0) ? HCUR : nullptr, (long long)N * 64);
  }

  // ---- classifier ----
  launch_gemm<64, 16, 64>(HCUR, 64, HCUR, 64, nullptr, P_fc, fc_b, (float*)d_out, 16, N,
                          0, stream);
}
